// A2aSparseMLP_34918084116586
// MI455X (gfx1250) — matrix operand order repaired and checked by reading, not compiled, not run on hardware
//
#include <hip/hip_runtime.h>

// ---------------- problem dims ----------------
#define T_TOK 8192
#define H_DIM 1024
#define E_NUM 16
#define I_DIM 512
#define N2_DIM 1024          // 2*INTER
#define SLOT_CAP 16640       // 2*T + 16*E padding

// ---------------- workspace layout (bytes) ----------------
static constexpr size_t OFF_TOPK_E    = 0;                            // int [T][2]
static constexpr size_t OFF_TOPK_S    = OFF_TOPK_E    + 65536;        // f32 [T][2]
static constexpr size_t OFF_ENTRY_OF  = OFF_TOPK_S    + 65536;        // int [T][2]  (e<<16|idx)
static constexpr size_t OFF_CNT       = OFF_ENTRY_OF  + 65536;        // int [16]
static constexpr size_t OFF_BASE      = OFF_CNT       + 256;          // int [17]
static constexpr size_t OFF_ENTRY_TOK = OFF_BASE      + 256;          // int [E][T]
static constexpr size_t OFF_XB        = OFF_ENTRY_TOK + 524288;       // bf16 [T][H]
static constexpr size_t OFF_WGU       = OFF_XB        + 16777216;     // bf16 [E][N2][H]
static constexpr size_t OFF_WDN       = OFF_WGU       + 33554432;     // bf16 [E][H][I]
static constexpr size_t OFF_ACT       = OFF_WDN       + 16777216;     // bf16 [SLOT_CAP][I]
static constexpr size_t OFF_DOUT      = OFF_ACT       + 17039360;     // f32  [SLOT_CAP][H]

typedef __attribute__((ext_vector_type(16))) __bf16 v16bf;
typedef __attribute__((ext_vector_type(8)))  __bf16 v8bf;
typedef __attribute__((ext_vector_type(8)))  float  v8f;

static __device__ __forceinline__ v16bf mk16(v8bf lo, v8bf hi) {
    v16bf r;
#pragma unroll
    for (int i = 0; i < 8; ++i) { r[i] = lo[i]; r[i + 8] = hi[i]; }
    return r;
}

static __device__ __forceinline__ v8f wmma_bf16(v16bf a, v16bf b, v8f c) {
    return __builtin_amdgcn_wmma_f32_16x16x32_bf16(false, a, false, b, (short)0, c, false, false);
}

// async global -> LDS copy of 16 bytes per lane (ASYNCcnt-tracked, GVS mode)
static __device__ __forceinline__ void ld_async_b128(const __bf16* lds_dst,
                                                     unsigned long long gbase,
                                                     unsigned byte_off) {
    unsigned lds_addr = (unsigned)(size_t)lds_dst;   // flat LDS addr truncates to LDS offset
    asm volatile("global_load_async_to_lds_b128 %0, %1, %2"
                 :: "v"(lds_addr), "v"(byte_off), "s"(gbase)
                 : "memory");
}
static __device__ __forceinline__ void wait_async() {
    asm volatile("s_wait_asynccnt 0" ::: "memory");
}

// ---------------- k0: zero expert counters ----------------
__global__ void k_zero_cnt(int* __restrict__ cnt) {
    if (threadIdx.x < E_NUM) cnt[threadIdx.x] = 0;
}

// ---------------- k1: x -> bf16 ----------------
__global__ void __launch_bounds__(256) k_cvt_x(const float* __restrict__ x,
                                               __bf16* __restrict__ xb, int n) {
    int i = blockIdx.x * 256 + threadIdx.x;
    if (i < n) xb[i] = (__bf16)x[i];
}

// ---------------- k2: tiled transpose + convert: src[e][K][N] f32 -> dst[e][N][K] bf16 -------
__global__ void __launch_bounds__(256) k_transpose_cvt(const float* __restrict__ src,
                                                       __bf16* __restrict__ dst,
                                                       int K, int N) {
    __shared__ float tile[32][33];
    const int e  = blockIdx.z;
    const int n0 = blockIdx.x * 32;
    const int k0 = blockIdx.y * 32;
    const int tx = threadIdx.x & 31;
    const int ty = threadIdx.x >> 5;  // 0..7
    const float* s = src + (size_t)e * K * N;
    __bf16*      d = dst + (size_t)e * K * N;
#pragma unroll
    for (int i = 0; i < 4; ++i) {
        int k = k0 + ty + i * 8;
        tile[ty + i * 8][tx] = s[(size_t)k * N + n0 + tx];
    }
    __syncthreads();
#pragma unroll
    for (int i = 0; i < 4; ++i) {
        int n = n0 + ty + i * 8;
        d[(size_t)n * K + k0 + tx] = (__bf16)tile[tx][ty + i * 8];
    }
}

// ---------------- k3: router (1 block / token) ----------------
__global__ void __launch_bounds__(256) k_router(const float* __restrict__ x,
                                                const float* __restrict__ rw,
                                                int*   __restrict__ topk_e,
                                                float* __restrict__ topk_s,
                                                int*   __restrict__ cnt,
                                                int*   __restrict__ entry_tok,
                                                int*   __restrict__ entry_of) {
    __shared__ float lacc[256 * E_NUM];
    const int t   = blockIdx.x;
    const int tid = threadIdx.x;
    float acc[E_NUM];
#pragma unroll
    for (int e = 0; e < E_NUM; ++e) acc[e] = 0.0f;
    for (int h = tid; h < H_DIM; h += 256) {
        float xv = x[(size_t)t * H_DIM + h];
        const float* r = rw + (size_t)h * E_NUM;
#pragma unroll
        for (int e = 0; e < E_NUM; ++e) acc[e] += xv * r[e];
    }
#pragma unroll
    for (int e = 0; e < E_NUM; ++e) lacc[tid * E_NUM + e] = acc[e];
    for (int s = 128; s >= 1; s >>= 1) {
        __syncthreads();
        for (int i = tid; i < s * E_NUM; i += 256) lacc[i] += lacc[i + s * E_NUM];
    }
    __syncthreads();
    if (tid == 0) {
        float l0 = -3.0e38f; int e0 = 0;
#pragma unroll
        for (int e = 0; e < E_NUM; ++e) { float v = lacc[e]; if (v > l0) { l0 = v; e0 = e; } }
        float l1 = -3.0e38f; int e1 = 0;
#pragma unroll
        for (int e = 0; e < E_NUM; ++e) { float v = lacc[e]; if (e != e0 && v > l1) { l1 = v; e1 = e; } }
        float w1  = __expf(l1 - l0);
        float inv = 1.0f / (1.0f + w1);
        float s0  = inv, s1 = w1 * inv;
        topk_e[t * 2 + 0] = e0;  topk_e[t * 2 + 1] = e1;
        topk_s[t * 2 + 0] = s0;  topk_s[t * 2 + 1] = s1;
        int i0 = atomicAdd(&cnt[e0], 1);
        int i1 = atomicAdd(&cnt[e1], 1);
        entry_tok[e0 * T_TOK + i0] = t;
        entry_tok[e1 * T_TOK + i1] = t;
        entry_of[t * 2 + 0] = (e0 << 16) | i0;
        entry_of[t * 2 + 1] = (e1 << 16) | i1;
    }
}

// ---------------- k4: exclusive scan of 16-padded counts ----------------
__global__ void k_scan(const int* __restrict__ cnt, int* __restrict__ base16) {
    if (threadIdx.x == 0) {
        int run = 0;
#pragma unroll
        for (int e = 0; e < E_NUM; ++e) {
            base16[e] = run;
            run += (cnt[e] + 15) & ~15;
        }
        base16[E_NUM] = run;
    }
}

// ---------------- k5: gate_up GEMM + GLU, B-stationary in LDS ----------------
// grid (E, 8, 16), block 256 (8 waves).
// Workgroup stages 64 gu-rows (= 32 act cols) x K=1024 of weights into LDS via
// async copies; each wave streams its own M-tiles (mt = mg + 8*wave + 64*i).
#define GU_RP 1032   // padded LDS row stride (elems)
__global__ void __launch_bounds__(256) k_gate_up(const __bf16* __restrict__ xb,
                                                 const __bf16* __restrict__ wgu,
                                                 const float*  __restrict__ gub,
                                                 const int*    __restrict__ cnt,
                                                 const int*    __restrict__ base16,
                                                 const int*    __restrict__ entry_tok,
                                                 __bf16*       __restrict__ act) {
    extern __shared__ __bf16 Bsh[];          // [64][GU_RP]
    const int e  = blockIdx.x;
    const int mg = blockIdx.y;               // 0..7
    const int nc = blockIdx.z;               // 0..15 : 32 act cols each
    const int c  = cnt[e];
    const int tiles = (c + 15) >> 4;
    if (mg >= tiles) return;

    // ---- stage B chunk: gu rows [nc*64, nc*64+64) x 1024 ----
    {
        const unsigned long long gb =
            (unsigned long long)(size_t)(wgu + (size_t)e * N2_DIM * H_DIM
                                             + (size_t)(nc * 64) * H_DIM);
        const int row = threadIdx.x >> 2;            // 0..63
        const int q   = threadIdx.x & 3;             // quarter of a row
        const unsigned g0 = (unsigned)((row * H_DIM + q * 256) * 2);
        const __bf16* l0 = &Bsh[row * GU_RP + q * 256];
#pragma unroll
        for (int i = 0; i < 256; i += 8)
            ld_async_b128(l0 + i, gb, g0 + i * 2);
    }
    wait_async();
    __syncthreads();

    const int wave = threadIdx.x >> 5;
    const int lane = threadIdx.x & 31;
    const int hl   = lane >> 4;
    const int ln   = lane & 15;
    const int acb  = nc * 32;                 // first act col of this block

    // lane-constant B fragment pointers (LDS)
    const __bf16* b0g = &Bsh[(2 * ln    ) * GU_RP + hl * 16];
    const __bf16* b0u = &Bsh[(2 * ln + 1) * GU_RP + hl * 16];
    const __bf16* b1g = &Bsh[(2 * (16 + ln)    ) * GU_RP + hl * 16];
    const __bf16* b1u = &Bsh[(2 * (16 + ln) + 1) * GU_RP + hl * 16];

    const float* be = gub + (size_t)e * N2_DIM;
    const float bgb0 = be[2 * (acb + ln)];
    const float bub0 = be[2 * (acb + ln) + 1];
    const float bgb1 = be[2 * (acb + 16 + ln)];
    const float bub1 = be[2 * (acb + 16 + ln) + 1];
    const int slotb = base16[e];

    for (int mt = mg + 8 * wave; mt < tiles; mt += 64) {
        int r = mt * 16 + ln;
        if (r >= c) r = mt * 16;
        const int tok = entry_tok[e * T_TOK + r];
        const __bf16* arow = xb + (size_t)tok * H_DIM + hl * 16;

        v8f ag0 = {}, au0 = {}, ag1 = {}, au1 = {};
        for (int kk = 0; kk < H_DIM; kk += 32) {
            v16bf a = mk16(*(const v8bf*)(arow + kk), *(const v8bf*)(arow + kk + 8));
            ag0 = wmma_bf16(a, mk16(*(const v8bf*)(b0g + kk), *(const v8bf*)(b0g + kk + 8)), ag0);
            au0 = wmma_bf16(a, mk16(*(const v8bf*)(b0u + kk), *(const v8bf*)(b0u + kk + 8)), au0);
            ag1 = wmma_bf16(a, mk16(*(const v8bf*)(b1g + kk), *(const v8bf*)(b1g + kk + 8)), ag1);
            au1 = wmma_bf16(a, mk16(*(const v8bf*)(b1u + kk), *(const v8bf*)(b1u + kk + 8)), au1);
        }

        const int slot = slotb + mt * 16;
#pragma unroll
        for (int j = 0; j < 8; ++j) {
            const int m = j + hl * 8;
            float g = fminf(ag0[j] + bgb0, 7.0f);
            float u = fminf(fmaxf(au0[j] + bub0, -7.0f), 7.0f);
            float glu = g / (1.0f + __expf(-1.702f * g));
            act[(size_t)(slot + m) * I_DIM + acb + ln] = (__bf16)((u + 1.0f) * glu);
            g   = fminf(ag1[j] + bgb1, 7.0f);
            u   = fminf(fmaxf(au1[j] + bub1, -7.0f), 7.0f);
            glu = g / (1.0f + __expf(-1.702f * g));
            act[(size_t)(slot + m) * I_DIM + acb + 16 + ln] = (__bf16)((u + 1.0f) * glu);
        }
    }
}

// ---------------- k6: down GEMM, B-stationary in LDS ----------------
// grid (E, 8, 16), block 256 (8 waves). LDS holds 64 h-rows x K=512.
#define DN_RP 520    // padded LDS row stride (elems)
__global__ void __launch_bounds__(256) k_down(const __bf16* __restrict__ act,
                                              const __bf16* __restrict__ wdn,
                                              const int*    __restrict__ cnt,
                                              const int*    __restrict__ base16,
                                              float*        __restrict__ dout) {
    extern __shared__ __bf16 Bsh[];          // [64][DN_RP]
    const int e  = blockIdx.x;
    const int mg = blockIdx.y;
    const int nc = blockIdx.z;               // 64 h cols each
    const int c  = cnt[e];
    const int tiles = (c + 15) >> 4;
    if (mg >= tiles) return;

    {
        const unsigned long long gb =
            (unsigned long long)(size_t)(wdn + (size_t)e * H_DIM * I_DIM
                                             + (size_t)(nc * 64) * I_DIM);
        const int row = threadIdx.x >> 2;
        const int q   = threadIdx.x & 3;             // 128 elems per quarter
        const unsigned g0 = (unsigned)((row * I_DIM + q * 128) * 2);
        const __bf16* l0 = &Bsh[row * DN_RP + q * 128];
#pragma unroll
        for (int i = 0; i < 128; i += 8)
            ld_async_b128(l0 + i, gb, g0 + i * 2);
    }
    wait_async();
    __syncthreads();

    const int wave = threadIdx.x >> 5;
    const int lane = threadIdx.x & 31;
    const int hl   = lane >> 4;
    const int ln   = lane & 15;
    const int slotb = base16[e];
    const int hbase = nc * 64;

    const __bf16* brow[4];
#pragma unroll
    for (int t = 0; t < 4; ++t)
        brow[t] = &Bsh[(t * 16 + ln) * DN_RP + hl * 16];

    for (int mt = mg + 8 * wave; mt < tiles; mt += 64) {
        const int slot = slotb + mt * 16;
        const __bf16* arow = act + (size_t)(slot + ln) * I_DIM + hl * 16;

        v8f accv[4] = {{}, {}, {}, {}};
        for (int kk = 0; kk < I_DIM; kk += 32) {
            v16bf a = mk16(*(const v8bf*)(arow + kk), *(const v8bf*)(arow + kk + 8));
#pragma unroll
            for (int t = 0; t < 4; ++t) {
                v16bf b = mk16(*(const v8bf*)(brow[t] + kk), *(const v8bf*)(brow[t] + kk + 8));
                accv[t] = wmma_bf16(a, b, accv[t]);
            }
        }
#pragma unroll
        for (int t = 0; t < 4; ++t)
#pragma unroll
            for (int j = 0; j < 8; ++j)
                dout[(size_t)(slot + j + hl * 8) * H_DIM + hbase + t * 16 + ln] = accv[t][j];
    }
}

// ---------------- k7: combine (1 block / token) ----------------
__global__ void __launch_bounds__(256) k_combine(const float* __restrict__ dnb,
                                                 const int*   __restrict__ entry_of,
                                                 const float* __restrict__ topk_s,
                                                 const int*   __restrict__ base16,
                                                 const float* __restrict__ dout,
                                                 float*       __restrict__ out) {
    const int t   = blockIdx.x;
    const int tid = threadIdx.x;
    const int v0 = entry_of[t * 2 + 0], v1 = entry_of[t * 2 + 1];
    const int e0 = v0 >> 16, i0 = v0 & 0xFFFF;
    const int e1 = v1 >> 16, i1 = v1 & 0xFFFF;
    const size_t s0r = (size_t)(base16[e0] + i0) * H_DIM;
    const size_t s1r = (size_t)(base16[e1] + i1) * H_DIM;
    const float s0 = topk_s[t * 2 + 0], s1 = topk_s[t * 2 + 1];
    for (int h = tid; h < H_DIM; h += 256) {
        float r = s0 * (dout[s0r + h] + dnb[(size_t)e0 * H_DIM + h]) +
                  s1 * (dout[s1r + h] + dnb[(size_t)e1 * H_DIM + h]);
        out[(size_t)t * H_DIM + h] = r;
    }
}

// ---------------- launch ----------------
extern "C" void kernel_launch(void* const* d_in, const int* in_sizes, int n_in,
                              void* d_out, int out_size, void* d_ws, size_t ws_size,
                              hipStream_t stream) {
    const float* x   = (const float*)d_in[0];
    const float* rw  = (const float*)d_in[1];
    const float* gup = (const float*)d_in[2];
    const float* gub = (const float*)d_in[3];
    const float* dnp = (const float*)d_in[4];
    const float* dnb = (const float*)d_in[5];
    float* out = (float*)d_out;

    char* ws = (char*)d_ws;
    int*    topk_e    = (int*)   (ws + OFF_TOPK_E);
    float*  topk_s    = (float*) (ws + OFF_TOPK_S);
    int*    entry_of  = (int*)   (ws + OFF_ENTRY_OF);
    int*    cnt       = (int*)   (ws + OFF_CNT);
    int*    base16    = (int*)   (ws + OFF_BASE);
    int*    entry_tok = (int*)   (ws + OFF_ENTRY_TOK);
    __bf16* xb        = (__bf16*)(ws + OFF_XB);
    __bf16* wgu       = (__bf16*)(ws + OFF_WGU);
    __bf16* wdn       = (__bf16*)(ws + OFF_WDN);
    __bf16* act       = (__bf16*)(ws + OFF_ACT);
    float*  dout      = (float*) (ws + OFF_DOUT);

    k_zero_cnt<<<1, 32, 0, stream>>>(cnt);

    const int nx = T_TOK * H_DIM;
    k_cvt_x<<<(nx + 255) / 256, 256, 0, stream>>>(x, xb, nx);

    k_transpose_cvt<<<dim3(N2_DIM / 32, H_DIM / 32, E_NUM), 256, 0, stream>>>(gup, wgu, H_DIM, N2_DIM);
    k_transpose_cvt<<<dim3(H_DIM / 32, I_DIM / 32, E_NUM), 256, 0, stream>>>(dnp, wdn, I_DIM, H_DIM);

    k_router<<<T_TOK, 256, 0, stream>>>(x, rw, topk_e, topk_s, cnt, entry_tok, entry_of);
    k_scan<<<1, 32, 0, stream>>>(cnt, base16);

    k_gate_up<<<dim3(E_NUM, 8, 16), 256, 64 * GU_RP * 2, stream>>>(xb, wgu, gub, cnt, base16, entry_tok, act);
    k_down   <<<dim3(E_NUM, 8, 16), 256, 64 * DN_RP * 2, stream>>>(act, wdn, cnt, base16, dout);
    k_combine<<<T_TOK, 256, 0, stream>>>(dnb, entry_of, topk_s, base16, dout, out);
}
